// GCN_66443144069641
// MI455X (gfx1250) — compile-verified, matrix operand order checked by reading, not run on hardware
//
#include <hip/hip_runtime.h>
#include <hip/hip_bf16.h>

typedef __attribute__((ext_vector_type(16))) _Float16 v16h;
typedef __attribute__((ext_vector_type(8)))  float    v8f;

// ---------------------------------------------------------------------------
// Degree accumulation: out_deg[src[e]] += 1, in_deg[dst[e]] += 1
// ---------------------------------------------------------------------------
__global__ void gcn_degree_kernel(const int* __restrict__ src,
                                  const int* __restrict__ dst,
                                  float* __restrict__ odeg,
                                  float* __restrict__ ideg, int E) {
    int e = blockIdx.x * blockDim.x + threadIdx.x;
    if (e < E) {
        unsafeAtomicAdd(&odeg[src[e]], 1.0f);
        unsafeAtomicAdd(&ideg[dst[e]], 1.0f);
    }
}

// deg -> rsqrt(max(deg,1)) in place, both arrays
__global__ void gcn_norm_kernel(float* __restrict__ a, float* __restrict__ b, int n) {
    int i = blockIdx.x * blockDim.x + threadIdx.x;
    if (i < n) {
        a[i] = rsqrtf(fmaxf(a[i], 1.0f));
        b[i] = rsqrtf(fmaxf(b[i], 1.0f));
    }
}

// ---------------------------------------------------------------------------
// SpMM scatter: agg[dst[e], :] += h[src[e], :] * out_norm[src[e]]
// One thread per (edge, 4-feature chunk). For F=128 a full wave32 covers one
// edge row -> coalesced float4 gather + global_atomic_add_f32 scatter.
// ---------------------------------------------------------------------------
template <int F>
__global__ void gcn_scatter_kernel(const float* __restrict__ h,
                                   const int* __restrict__ src,
                                   const int* __restrict__ dst,
                                   const float* __restrict__ onorm,
                                   float* __restrict__ agg, int E) {
    constexpr int F4 = F / 4;
    int idx = blockIdx.x * blockDim.x + threadIdx.x;
    if (idx >= E * F4) return;
    int e = idx / F4;
    int c = (idx - e * F4) * 4;
    int s = src[e];
    int d = dst[e];
    float sc = onorm[s];
    float4 v = *(const float4*)(h + (size_t)s * F + c);
    float* ap = agg + (size_t)d * F + c;
    unsafeAtomicAdd(ap + 0, v.x * sc);
    unsafeAtomicAdd(ap + 1, v.y * sc);
    unsafeAtomicAdd(ap + 2, v.z * sc);
    unsafeAtomicAdd(ap + 3, v.w * sc);
}

// ---------------------------------------------------------------------------
// Dense GEMM with WMMA: out[N,K] = relu?((A[N,F] * innorm[:,None]) @ W[F,K] + b)
// One wave32 per 16-row panel; it computes ALL K/16 column tiles so each A
// fragment is loaded once per k-step and reused K/16 times (4x less A/W
// traffic than one-tile-per-wave). K-loop in steps of 32 using
// v_wmma_f32_16x16x32_f16 (f16 A/B, f32 accumulate).
//
// Fragment layouts per CDNA5 ISA 7.12.2 (wave32):
//   A 16x32 f16 : lane L -> row M=L&15; half-K base = 8*(L>>4);
//                 v16h elems 0..7 -> K=kb..kb+7, elems 8..15 -> K=kb+16..kb+23
//   B 32x16 f16 : symmetric, lane L -> col N=L&15, same K striping
//   C/D 16x16 f32: lane L -> col N=L&15; elem v -> row M = v + 8*(L>>4)
// Requires N % 16 == 0, K % 16 == 0, F % 32 == 0 (all true here).
// ---------------------------------------------------------------------------
template <int F, int K, bool RELU>
__global__ void __launch_bounds__(32)
gcn_gemm_wmma_kernel(const float* __restrict__ A,
                     const float* __restrict__ innorm,
                     const float* __restrict__ W,
                     const float* __restrict__ bias,
                     float* __restrict__ out) {
    constexpr int NT = K / 16;                // column tiles per wave (<=4)
    const int lane  = threadIdx.x;            // 0..31, wave32
    const int mr    = lane & 15;
    const int khalf = (lane >> 4) << 3;       // 0 or 8
    const int m0    = blockIdx.x * 16;

    const int   arow  = m0 + mr;              // row this lane loads for A
    const float scale = innorm[arow];
    const float* __restrict__ Ap = A + (size_t)arow * F;

    v8f acc[NT] = {};

#pragma unroll
    for (int k0 = 0; k0 < F; k0 += 32) {
        const int kb = k0 + khalf;
        v16h af;
#pragma unroll
        for (int j = 0; j < 8; ++j) {
            af[j]     = (_Float16)(Ap[kb + j] * scale);
            af[8 + j] = (_Float16)(Ap[kb + 16 + j] * scale);
        }
#pragma unroll
        for (int t = 0; t < NT; ++t) {
            const int bcol = t * 16 + mr;     // B column this lane loads
            v16h bf;
#pragma unroll
            for (int j = 0; j < 8; ++j) {
                bf[j]     = (_Float16)(W[(kb + j) * K + bcol]);
                bf[8 + j] = (_Float16)(W[(kb + 16 + j) * K + bcol]);
            }
            acc[t] = __builtin_amdgcn_wmma_f32_16x16x32_f16(
                /*neg_a=*/false, af, /*neg_b=*/false, bf,
                /*c_mod=*/(short)0, acc[t], /*reuse_a=*/false, /*reuse_b=*/false);
        }
    }

#pragma unroll
    for (int t = 0; t < NT; ++t) {
        const int bcol = t * 16 + mr;
        const float bv = bias[bcol];
#pragma unroll
        for (int v = 0; v < 8; ++v) {
            int mrow = m0 + v + khalf;        // M = v + 8*(lane>>4)
            float val = acc[t][v] + bv;
            if (RELU) val = fmaxf(val, 0.0f);
            out[(size_t)mrow * K + bcol] = val;
        }
    }
}

// ---------------------------------------------------------------------------
// Orchestration
// inputs: x, W0, b0, W1, b1, W2, b2, src, dst
// ---------------------------------------------------------------------------
extern "C" void kernel_launch(void* const* d_in, const int* in_sizes, int n_in,
                              void* d_out, int out_size, void* d_ws, size_t ws_size,
                              hipStream_t stream) {
    const float* x  = (const float*)d_in[0];
    const float* W0 = (const float*)d_in[1];
    const float* b0 = (const float*)d_in[2];
    const float* W1 = (const float*)d_in[3];
    const float* b1 = (const float*)d_in[4];
    const float* W2 = (const float*)d_in[5];
    const float* b2 = (const float*)d_in[6];
    const int* src  = (const int*)d_in[7];
    const int* dst  = (const int*)d_in[8];
    float* out = (float*)d_out;

    const int IN_DIM = 128, HID = 64, NCLS = 16;
    const int N = in_sizes[0] / IN_DIM;   // 100000 (multiple of 16)
    const int E = in_sizes[7];            // 1600000

    // workspace layout (floats)
    float* ws    = (float*)d_ws;
    float* onorm = ws;                        // N
    float* inorm = ws + N;                    // N
    float* agg   = ws + 2 * (size_t)N;        // N * 128 (reused, later N*64)
    float* h     = agg + (size_t)N * IN_DIM;  // N * 64  (reused h1/h2)

    const int T = 256;

    // degrees -> norms
    hipMemsetAsync(onorm, 0, 2 * (size_t)N * sizeof(float), stream);
    gcn_degree_kernel<<<(E + T - 1) / T, T, 0, stream>>>(src, dst, onorm, inorm, E);
    gcn_norm_kernel<<<(N + T - 1) / T, T, 0, stream>>>(onorm, inorm, N);

    // ---- layer 0: F=128 -> HID=64, relu ----
    hipMemsetAsync(agg, 0, (size_t)N * IN_DIM * sizeof(float), stream);
    {
        int nthreads = E * (IN_DIM / 4);
        gcn_scatter_kernel<128><<<(nthreads + T - 1) / T, T, 0, stream>>>(
            x, src, dst, onorm, agg, E);
        gcn_gemm_wmma_kernel<128, 64, true><<<dim3(N / 16), 32, 0, stream>>>(
            agg, inorm, W0, b0, h);
    }

    // ---- layer 1: F=64 -> HID=64, relu ----
    hipMemsetAsync(agg, 0, (size_t)N * HID * sizeof(float), stream);
    {
        int nthreads = E * (HID / 4);
        gcn_scatter_kernel<64><<<(nthreads + T - 1) / T, T, 0, stream>>>(
            h, src, dst, onorm, agg, E);
        gcn_gemm_wmma_kernel<64, 64, true><<<dim3(N / 16), 32, 0, stream>>>(
            agg, inorm, W1, b1, h);
    }

    // ---- layer 2: F=64 -> NCLS=16, no relu ----
    hipMemsetAsync(agg, 0, (size_t)N * HID * sizeof(float), stream);
    {
        int nthreads = E * (HID / 4);
        gcn_scatter_kernel<64><<<(nthreads + T - 1) / T, T, 0, stream>>>(
            h, src, dst, onorm, agg, E);
        gcn_gemm_wmma_kernel<64, 16, false><<<dim3(N / 16), 32, 0, stream>>>(
            agg, inorm, W2, b2, out);
    }
}
